// SpectralConv1d_76115410420342
// MI455X (gfx1250) — matrix operand order repaired and checked by reading, not compiled, not run on hardware
//
#include <hip/hip_runtime.h>
#include <stdint.h>

// ---------------------------------------------------------------------------
// SpectralConv1d fused kernel for gfx1250 (MI455X).
//   Stage A: Y = Fs(32x32) @ x        (rfft as small GEMM)        -> LDS
//   Stage B: Z = Y @ Wbig(256x256)    (per-mode complex mix)      -> LDS
//   Stage C: out = G(32x32) @ Z       (irfft as small GEMM)       -> HBM
// All GEMMs use v_wmma_f32_16x16x32_bf16. x tile staged via TDM.
// ---------------------------------------------------------------------------

#define T_LEN   32
#define NCOLS   32768
#define CIN     128
#define COUT    128
#define MODES   16
#define NT      16          // n-columns per workgroup
#define KSTEPS  8           // 256 / 32 K-steps in stage B

typedef __attribute__((ext_vector_type(16))) __bf16        v16bf;
typedef __attribute__((ext_vector_type(8)))  float         v8f;
typedef __attribute__((ext_vector_type(4)))  unsigned int  v4u;
typedef __attribute__((ext_vector_type(8)))  int           v8i;
typedef __attribute__((ext_vector_type(4)))  int           v4i;

union Frag {
    v16bf          v;
    __bf16         b[16];
    unsigned short u[16];
    v4u            q[2];
};

// Native f32 -> bf16 convert (single v_cvt, RNE) instead of bit twiddling.
__device__ __forceinline__ __bf16 f2bf(float f) { return (__bf16)f; }

// A-operand k order used by this code (contiguous, not AMD's table order): lanes 0-15 hold K 0-15, lanes
// 16-31 hold K 16-31; row M = lane%16.
__device__ __forceinline__ int kmapA(int lane, int e) {
    return ((lane & 16) ? 16 : 0) + e;
}
// B-operand k order used here (contiguous, same as A): lanes 0-15 hold K 0-15, lanes 16-31 K 16-31;
// column N = lane%16.
__device__ __forceinline__ int kmapB(int lane, int e) {
    return ((lane & 16) ? 16 : 0) + e;
}

#define WMMA_BF16(A, B, C) \
    __builtin_amdgcn_wmma_f32_16x16x32_bf16(false, (A), false, (B), (short)0, (C), false, false)

#ifdef __has_builtin
#if __has_builtin(__builtin_amdgcn_tensor_load_to_lds)
#define HAVE_TDM 1
#endif
#if __has_builtin(__builtin_amdgcn_s_wait_tensorcnt)
#define HAVE_TCNT 1
#endif
#endif

// ---------------------------------------------------------------------------
// Prep kernel: build stacked real mixing matrix Wbig[m][k=256][j=256] in bf16,
// pre-swizzled into per-lane WMMA B-fragment order so the main kernel streams
// operands with two coalesced b128 loads per lane.
//   k<128 : Y_re(i=k)      k>=128 : Y_im(i=k-128)
//   j<128 : Z_re(o=j)      j>=128 : Z_im(o=j-128)
//   [Zre Zim] = [Yre Yim] @ [[Wre, Wim], [-Wim, Wre]]
// frag element order: idx = (((m*8+kt)*16+jt)*32+lane)*16 + e
// ---------------------------------------------------------------------------
__global__ __launch_bounds__(256) void prep_wfrag(const float* __restrict__ w,
                                                  __bf16* __restrict__ wf) {
    int g    = blockIdx.x * blockDim.x + threadIdx.x;   // 0 .. 65535
    int lane = g & 31;
    int jt   = (g >> 5) & 15;
    int kt   = (g >> 9) & 7;
    int m    = (g >> 12) & 15;

    int  j   = jt * 16 + (lane & 15);
    int  o   = j & 127;
    bool zre = (j < 128);

    __bf16* dst = wf + (size_t)g * 16;
#pragma unroll
    for (int e = 0; e < 16; ++e) {
        int  K   = kt * 32 + kmapB(lane, e);
        int  i   = K & 127;
        bool xre = (K < 128);
        const float* wp = w + (((size_t)i * COUT + o) * MODES + m) * 2;
        float wre = wp[0];
        float wim = wp[1];
        float v   = zre ? (xre ? wre : -wim) : (xre ? wim : wre);
        dst[e] = f2bf(v);
    }
}

// ---------------------------------------------------------------------------
// Main fused kernel. One workgroup = 16 n-columns, 8 waves.
// Dynamic LDS (256 KB):
//   [0      , 128K) : sY  : Y[s=32][n=16][i=128] bf16   (i contiguous)
//   [128K   , 256K) : scratch: stage A x-tile f32 [t=32][n=16][i=64] (per half)
//                     then reused as Z[n=16][o=128][s=32] bf16 (s contiguous,
//                     so stage C K-operands are two ds_load_b128 per lane)
// ---------------------------------------------------------------------------
__global__ __launch_bounds__(256) void spectral_fused(const float* __restrict__ x,
                                                      const __bf16* __restrict__ wfrag,
                                                      float* __restrict__ out) {
    extern __shared__ __align__(16) unsigned char smem[];
    __bf16* sY   = (__bf16*)smem;            // 128 KB
    float*  xbuf = (float*)(smem + 131072);  // 128 KB (stage A)
    __bf16* zbuf = (__bf16*)(smem + 131072); // reuse   (stage B/C)

    const int tid  = threadIdx.x;
    const int lane = tid & 31;
    const int wave = tid >> 5;
    const int n0   = blockIdx.x * NT;
    const int l15  = lane & 15;
    const int rhi  = (lane & 16) ? 8 : 0;   // D-matrix row offset for lanes 16-31

    // ---- per-lane twiddle A-fragments (registers) --------------------------
    const float TWO_PI_32 = 6.2831853071795864769f / 32.0f;
    Frag aC, aS, gLo, gHi;
#pragma unroll
    for (int e = 0; e < 16; ++e) {
        int t  = kmapA(lane, e);
        int ph = (l15 * t) & 31;
        aC.b[e] = f2bf(__cosf(TWO_PI_32 * (float)ph));   // rfft real rows
        aS.b[e] = f2bf(-__sinf(TWO_PI_32 * (float)ph));  // rfft imag rows
        // irfft matrix G[t][s]
        int s = kmapA(lane, e);
        float glo, ghi;
        if (s < 16) {
            float c = (s == 0) ? (1.0f / 32.0f) : (1.0f / 16.0f);
            glo = c * __cosf(TWO_PI_32 * (float)((s * l15) & 31));
            ghi = c * __cosf(TWO_PI_32 * (float)((s * (l15 + 16)) & 31));
        } else if (s == 16) {
            glo = 0.0f; ghi = 0.0f;                       // imag of DC bin ignored
        } else {
            int mm = s - 16;
            glo = -(1.0f / 16.0f) * __sinf(TWO_PI_32 * (float)((mm * l15) & 31));
            ghi = -(1.0f / 16.0f) * __sinf(TWO_PI_32 * (float)((mm * (l15 + 16)) & 31));
        }
        gLo.b[e] = f2bf(glo);
        gHi.b[e] = f2bf(ghi);
    }

    // ======================= Stage A: Y = Fs @ x ============================
    for (int half = 0; half < 2; ++half) {
        const int ibase = half * 64;
#ifdef HAVE_TDM
        if (wave == 0) {
            // 3-D tile: dim0=i(64, contig), dim1=n(16, stride 128),
            //           dim2=t(32, stride N*128); f32 elements.
            unsigned long long ga = (unsigned long long)(uintptr_t)x +
                                    4ull * ((unsigned long long)n0 * CIN + ibase);
            unsigned int ldsa = (unsigned int)(uintptr_t)xbuf;  // LDS byte address
            v4u g0;
            g0[0] = 1u;                                   // count=1, user descriptor
            g0[1] = ldsa;                                 // lds_addr
            g0[2] = (unsigned int)ga;                     // global_addr[31:0]
            g0[3] = ((unsigned int)(ga >> 32) & 0x1FFFFFFu) | (2u << 30); // type=2
            v8i g1;
            unsigned int d0 = 1u << 30, d1 = 1u << 30;    // huge tensor dims: no OOB
            g1[0] = (int)(2u << 16);                      // data_size=4B
            g1[1] = (int)((d0 & 0xFFFFu) << 16);          // tensor_dim0[15:0]
            g1[2] = (int)(((d0 >> 16) & 0xFFFFu) | ((d1 & 0xFFFFu) << 16));
            g1[3] = (int)(((d1 >> 16) & 0xFFFFu) | (64u << 16));   // tile_dim0=64
            g1[4] = (int)(16u | (32u << 16));             // tile_dim1=16, tile_dim2=32
            g1[5] = (int)128u;                            // dim0_stride = 128 elems
            g1[6] = 0;                                    // dim0_str[47:32]|dim1_str[15:0]
            g1[7] = (int)((NCOLS * CIN) >> 16);           // dim1_stride = N*128
            v4i g2; g2[0] = (int)(1u << 30); g2[1] = 1; g2[2] = 0; g2[3] = 0;
            v4i g3; g3[0] = 0; g3[1] = 0; g3[2] = 0; g3[3] = 0;
            v8i g4; g4[0] = 0; g4[1] = 0; g4[2] = 0; g4[3] = 0;
                    g4[4] = 0; g4[5] = 0; g4[6] = 0; g4[7] = 0;
            __builtin_amdgcn_tensor_load_to_lds(g0, g1, g2, g3, g4, 0);
#ifdef HAVE_TCNT
            __builtin_amdgcn_s_wait_tensorcnt(0);
#else
            asm volatile("s_wait_tensorcnt 0x0" ::: "memory");
#endif
        }
#else
        // Fallback: cooperative coalesced copy global -> LDS
        for (int l = tid; l < 32 * NT * 64; l += 256) {
            int i = l & 63, n = (l >> 6) & 15, t = l >> 10;
            xbuf[l] = x[((size_t)t * NCOLS + n0 + n) * CIN + ibase + i];
        }
#endif
        __syncthreads();

        // 64 column tiles (16 n x 4 i-chunks); 8 per wave.
        for (int ct = wave; ct < 64; ct += 8) {
            int n  = ct >> 2;
            int ic = (ct & 3) * 16;
            Frag b;
#pragma unroll
            for (int e = 0; e < 16; ++e) {
                int t = kmapB(lane, e);
                b.b[e] = f2bf(xbuf[(t * NT + n) * 64 + ic + l15]);
            }
            v8f c0 = {}, c1 = {};
            c0 = WMMA_BF16(aC.v, b.v, c0);   // rows s = 0..15  (Re)
            c1 = WMMA_BF16(aS.v, b.v, c1);   // rows s = 16..31 (Im)
#pragma unroll
            for (int v = 0; v < 8; ++v) {
                int sr = v + rhi;
                int ii = ibase + ic + l15;
                sY[((sr)      * NT + n) * CIN + ii] = f2bf(c0[v]);
                sY[((sr + 16) * NT + n) * CIN + ii] = f2bf(c1[v]);
            }
        }
        __syncthreads();
    }

    // ==================== Stage B: Z = Y @ Wbig (per mode) ==================
    // Each wave owns j-tiles {2w, 2w+1} of 16 (j = stacked re|im output).
    const int jt0 = wave * 2;
    for (int m = 0; m < MODES; ++m) {
        // prefetch next mode's fragments for this wave (L2 temporal)
        if (m + 1 < MODES)
            __builtin_prefetch(wfrag + ((size_t)((m + 1) * 8 * 16 + jt0) * 32) * 16, 0, 1);

        v8f z0 = {}, z1 = {};
#pragma unroll
        for (int kt = 0; kt < KSTEPS; ++kt) {
            // A fragment from Y: row = n (lane%16), K = kt*32 + kmapA
            Frag a;
            int kb0 = kt * 32 + ((lane & 16) ? 16 : 0);     // K of elements 0-7
            int kb1 = kb0 + 8;                               // K of elements 8-15
            int s0  = m + ((kb0 >= 128) ? 16 : 0), i0 = kb0 & 127;
            int s1  = m + ((kb1 >= 128) ? 16 : 0), i1 = kb1 & 127;
            a.q[0] = *(const v4u*)&sY[(s0 * NT + l15) * CIN + i0];
            a.q[1] = *(const v4u*)&sY[(s1 * NT + l15) * CIN + i1];

            // B fragments: pre-swizzled, 2x b128 per lane, fully coalesced
            const v4u* w0 = (const v4u*)(wfrag + ((size_t)(((m * 8 + kt) * 16 + jt0)     * 32 + lane)) * 16);
            const v4u* w1 = (const v4u*)(wfrag + ((size_t)(((m * 8 + kt) * 16 + jt0 + 1) * 32 + lane)) * 16);
            Frag b0, b1;
            b0.q[0] = w0[0]; b0.q[1] = w0[1];
            b1.q[0] = w1[0]; b1.q[1] = w1[1];

            z0 = WMMA_BF16(a.v, b0.v, z0);
            z1 = WMMA_BF16(a.v, b1.v, z1);
        }
        // Store Z tiles -> zbuf[n][o][s]  (s contiguous for stage C)
#pragma unroll
        for (int v = 0; v < 8; ++v) {
            int nr = v + rhi;
            int sa = m + ((jt0     >= 8) ? 16 : 0);
            int sb = m + ((jt0 + 1 >= 8) ? 16 : 0);
            int oa = ((jt0)     & 7) * 16 + l15;
            int ob = ((jt0 + 1) & 7) * 16 + l15;
            zbuf[(nr * COUT + oa) * 32 + sa] = f2bf(z0[v]);
            zbuf[(nr * COUT + ob) * 32 + sb] = f2bf(z1[v]);
        }
    }
    __syncthreads();

    // ======================= Stage C: out = G @ Z ===========================
    // 128 column tiles (16 n x 8 o-chunks); 16 per wave.
    for (int ct = wave; ct < 128; ct += 8) {
        int n  = ct >> 3;
        int oc = (ct & 7) * 16;
        // B fragment: 16 consecutive s per lane -> two ds_load_b128
        Frag b;
        const __bf16* zp = &zbuf[(n * COUT + oc + l15) * 32 + ((lane & 16) ? 16 : 0)];
        b.q[0] = *(const v4u*)(zp);
        b.q[1] = *(const v4u*)(zp + 8);

        v8f d0 = {}, d1 = {};
        d0 = WMMA_BF16(gLo.v, b.v, d0);   // t = 0..15
        d1 = WMMA_BF16(gHi.v, b.v, d1);   // t = 16..31
#pragma unroll
        for (int v = 0; v < 8; ++v) {
            int tr = v + rhi;
            size_t base = ((size_t)(n0 + n)) * CIN + oc + l15;
            out[(size_t)tr        * NCOLS * CIN + base] = d0[v];
            out[(size_t)(tr + 16) * NCOLS * CIN + base] = d1[v];
        }
    }
}

// ---------------------------------------------------------------------------
extern "C" void kernel_launch(void* const* d_in, const int* in_sizes, int n_in,
                              void* d_out, int out_size, void* d_ws, size_t ws_size,
                              hipStream_t stream) {
    const float* x = (const float*)d_in[0];
    const float* w = (const float*)d_in[1];
    float* out = (float*)d_out;
    __bf16* wf = (__bf16*)d_ws;   // needs 2 MB

    (void)in_sizes; (void)n_in; (void)out_size; (void)ws_size;

    // 1) swizzle weights into WMMA B-fragment order (16*8*16*32 lane-frags)
    prep_wfrag<<<256, 256, 0, stream>>>(w, wf);

    // 2) fused spectral conv; 256 KB dynamic LDS per workgroup
    const int smem_bytes = 262144;
    (void)hipFuncSetAttribute(reinterpret_cast<const void*>(spectral_fused),
                              hipFuncAttributeMaxDynamicSharedMemorySize, smem_bytes);
    spectral_fused<<<dim3(NCOLS / NT), dim3(256), smem_bytes, stream>>>(x, wf, out);
}
